// Conv_55637006352855
// MI455X (gfx1250) — matrix operand order repaired and checked by reading, not compiled, not run on hardware
//
#include <hip/hip_runtime.h>

// Implicit-GEMM 3x3 conv (B=32,C=128,H=W=56 -> O=256), bf16 WMMA, f32 accum.
// M = O = 256, K = C*9 = 1152, N = B*H*W = 100352.
// K tiled as 2 macro-steps of 576 = 18*32 (multiple of 9): the 18-step inner
// loop is fully unrolled so (a) the im2col split k -> (c,kh,kw) constant-folds
// into global_load immediates and (b) the LDS ping-pong parity is a
// compile-time constant. Waves 0-3 gather B (im2col) through addrspace(1)
// pointers (global_load, LOADcnt only); waves 4-7 stream A (weights). All
// waves compute. Pipeline per step: stage slice s+1 -> LDS buf[(s+1)&1],
// fetch slice s+2 -> regs, compute buf[s&1], ONE barrier -- LDS stores and
// HBM loads overlap the WMMAs. Output stores are non-temporal so the
// write-once 103MB output doesn't evict the L2-resident input.

#define BM     128
#define BN     128
#define BK     32
#define LDK    40        // BK + 8 shorts pad -> 80B row stride, 16B-aligned rows
#define KTOT   1152
#define KMAC2  576       // 18 * 32, multiple of 9
#define C_IN   128
#define H_IN   56
#define W_IN   56
#define O_OUT  256
#define HW     3136      // 56*56
#define BATCH  32

typedef __attribute__((ext_vector_type(16))) __bf16 v16bf;
typedef __attribute__((ext_vector_type(8)))  float  v8f;

// Global (addrspace 1) pointer type: forces global_load instead of flat_load.
typedef const float __attribute__((address_space(1)))* gfp;

union FragBF {
    v16bf v;
    unsigned int u[8];
};

// Pack two fp32 -> packed bf16x2 (round-half-up) with a single v_perm_b32.
__device__ __forceinline__ unsigned int pkbf(float lo, float hi) {
    unsigned int a = __builtin_bit_cast(unsigned int, lo) + 0x8000u;
    unsigned int b = __builtin_bit_cast(unsigned int, hi) + 0x8000u;
    return __builtin_amdgcn_perm(b, a, 0x07060302u);
}

__global__ __launch_bounds__(256) void conv_igemm_bf16_wmma(
    const float* __restrict__ x, const float* __restrict__ w,
    const float* __restrict__ bias, float* __restrict__ out) {
    // Ping-pong LDS: 2 x (A 10KB + B 10KB) = 40KB
    __shared__ __align__(16) unsigned short Alds[2][BM][LDK];
    __shared__ __align__(16) unsigned short Blds[2][BN][LDK];

    const int tid      = threadIdx.x;
    const int lane     = tid & 31;
    const int laneHalf = lane >> 4;
    const int lane15   = lane & 15;
    const int wave     = tid >> 5;        // 8 waves
    const int waveM    = wave >> 1;       // 0..3 : 32-wide M slab
    const int waveN    = wave & 1;        // 0..1 : 64-wide N slab

    const int blockO = blockIdx.y * BM;
    const int blockN = blockIdx.x * BN;

    const int  fr  = tid & 127;           // fill row (0..127)
    const bool isB = tid < 128;           // waves 0-3: B gather; 4-7: A stream

    // ---- B-filler constants: pixel -> (batch, oh, ow), 9 validity masks ----
    const int gn = blockN + fr;
    const int bb = gn / HW;
    const int p0 = gn - bb * HW;
    const int oh = p0 / W_IN;
    const int ow = p0 - oh * W_IN;

    bool vm[9];
#pragma unroll
    for (int r9 = 0; r9 < 9; ++r9) {
        int kh = r9 / 3, kw = r9 - kh * 3;
        vm[r9] = (((unsigned)(oh + kh - 1) < (unsigned)H_IN) &
                  ((unsigned)(ow + kw - 1) < (unsigned)W_IN));
    }
    // Base points at (c=cBase, kh=0, kw=0) for this pixel (may be "off-grid";
    // per-element immediates add c*HW + kh*56 + kw, which restores validity).
    const float* pValid = x + (bb * C_IN) * HW + (oh - 1) * W_IN + (ow - 1);
    const float* pA     = w + (blockO + fr) * KTOT;

    v8f acc[2][4];
#pragma unroll
    for (int i = 0; i < 2; ++i)
#pragma unroll
        for (int j = 0; j < 4; ++j)
            acc[i][j] = (v8f)0.0f;

    float regs[32];                        // staged K-slice (32 elements)
    gfp ptrArr[9];                         // per-r9 clamped global base pointers

#define BUILD_PTRS()                                                         \
    {                                                                        \
        _Pragma("unroll")                                                    \
        for (int r9 = 0; r9 < 9; ++r9)                                       \
            ptrArr[r9] =                                                     \
                (gfp)(unsigned long long)(vm[r9] ? pValid : x);              \
    }

#define FETCH_B(d0)                                                          \
    {                                                                        \
        _Pragma("unroll")                                                    \
        for (int e = 0; e < 32; ++e) {                                       \
            const int d  = (d0) + e;       /* compile-time */                \
            const int c  = d / 9;                                            \
            const int r9 = d - c * 9;                                        \
            const int kh = r9 / 3;                                           \
            const int kw = r9 - kh * 3;                                      \
            float v = ptrArr[r9][c * HW + kh * W_IN + kw];                   \
            regs[e] = vm[r9] ? v : 0.0f;                                     \
        }                                                                    \
    }

#define FETCH_A(d0)                                                          \
    {                                                                        \
        _Pragma("unroll")                                                    \
        for (int q = 0; q < 8; ++q) {                                        \
            float4 f = *(const float4*)(pA + (d0) + q * 4);                  \
            regs[q * 4 + 0] = f.x; regs[q * 4 + 1] = f.y;                    \
            regs[q * 4 + 2] = f.z; regs[q * 4 + 3] = f.w;                    \
        }                                                                    \
    }

#define STAGE(par)                                                           \
    {                                                                        \
        unsigned short (*dst)[LDK] = isB ? Blds[par] : Alds[par];            \
        _Pragma("unroll")                                                    \
        for (int q2 = 0; q2 < 4; ++q2) {                                     \
            uint4 pk;                                                        \
            pk.x = pkbf(regs[q2 * 8 + 0], regs[q2 * 8 + 1]);                 \
            pk.y = pkbf(regs[q2 * 8 + 2], regs[q2 * 8 + 3]);                 \
            pk.z = pkbf(regs[q2 * 8 + 4], regs[q2 * 8 + 5]);                 \
            pk.w = pkbf(regs[q2 * 8 + 6], regs[q2 * 8 + 7]);                 \
            *(uint4*)&dst[fr][q2 * 8] = pk;                                  \
        }                                                                    \
    }

    // ---- prologue: slice 0 -> buf0, fetch slice 1 ----
    if (isB) {
        BUILD_PTRS();
        FETCH_B(0);
    } else {
        FETCH_A(0);
    }
    STAGE(0);
    __syncthreads();
    if (isB) FETCH_B(32)
    else     FETCH_A(32)

    for (int mk2 = 0; mk2 < 2; ++mk2) {
#pragma unroll
        for (int u = 0; u < 18; ++u) {     // global step s = mk2*18 + u
            // ---- stage slice s+1 into buf[(s+1)&1] (overlaps compute) ----
            if (mk2 == 0 || u < 17) STAGE((u + 1) & 1);

            // ---- prefetch slice s+2 while this step computes ----
            // d0 = (u+2)*32 relative to current macro base; at u >= 16 this
            // reaches into the next macro-step's channels (still in range).
            if (mk2 == 0 || u < 16) {
                if (isB) FETCH_B((u + 2) * 32)
                else     FETCH_A((u + 2) * 32)
            }

            // ---- compute buf[s&1]: 12x ds_load_b128 + 8 WMMAs ----
            {
                FragBF a[2], b[4];
#pragma unroll
                for (int tm = 0; tm < 2; ++tm) {
                    int row = waveM * 32 + tm * 16 + lane15;   // A: M = lane&15
                    *(uint4*)&a[tm].u[0] =
                        *(const uint4*)&Alds[u & 1][row][laneHalf * 8];
                    *(uint4*)&a[tm].u[4] =
                        *(const uint4*)&Alds[u & 1][row][16 + laneHalf * 8];
                }
#pragma unroll
                for (int tn = 0; tn < 4; ++tn) {
                    int row = waveN * 64 + tn * 16 + lane15;   // B: N = lane&15
                    *(uint4*)&b[tn].u[0] =
                        *(const uint4*)&Blds[u & 1][row][laneHalf * 8];
                    *(uint4*)&b[tn].u[4] =
                        *(const uint4*)&Blds[u & 1][row][16 + laneHalf * 8];
                }
#pragma unroll
                for (int tm = 0; tm < 2; ++tm)
#pragma unroll
                    for (int tn = 0; tn < 4; ++tn)
                        acc[tm][tn] = __builtin_amdgcn_wmma_f32_16x16x32_bf16(
                            false, a[tm].v, false, b[tn].v,
                            (short)0, acc[tm][tn], false, false);
            }

            __syncthreads();               // single barrier per 32-K step
        }
        // advance to next macro-step: cBase += 64
        pValid += 64 * HW;
        pA     += KMAC2;
        if (isB && mk2 == 0) BUILD_PTRS();
    }

    // ---- epilogue: bias hoisted per output channel, NT grouped stores ----
#pragma unroll
    for (int tm = 0; tm < 2; ++tm) {
#pragma unroll
        for (int j = 0; j < 8; ++j) {
            int   o  = blockO + waveM * 32 + tm * 16 + laneHalf * 8 + j;
            float bv = bias[o];
#pragma unroll
            for (int tn = 0; tn < 4; ++tn) {
                int n  = blockN + waveN * 64 + tn * 16 + lane15;
                int b2 = n / HW;
                int p2 = n - b2 * HW;
                // write-once output: non-temporal, don't evict L2-resident x
                __builtin_nontemporal_store(acc[tm][tn][j] + bv,
                                            &out[(b2 * O_OUT + o) * HW + p2]);
            }
        }
    }
#undef BUILD_PTRS
#undef FETCH_B
#undef FETCH_A
#undef STAGE
}

extern "C" void kernel_launch(void* const* d_in, const int* in_sizes, int n_in,
                              void* d_out, int out_size, void* d_ws, size_t ws_size,
                              hipStream_t stream) {
    (void)in_sizes; (void)n_in; (void)out_size; (void)d_ws; (void)ws_size;
    const float* x    = (const float*)d_in[0];
    const float* w    = (const float*)d_in[1];
    const float* bias = (const float*)d_in[2];
    float* out        = (float*)d_out;

    dim3 grid((BATCH * HW) / BN, O_OUT / BM);   // (784, 2)
    conv_igemm_bf16_wmma<<<grid, dim3(256, 1, 1), 0, stream>>>(x, w, bias, out);
}
